// GPTBlock_15788299780318
// MI455X (gfx1250) — matrix operand order repaired and checked by reading, not compiled, not run on hardware
//
#include <hip/hip_runtime.h>

// ---------------------------------------------------------------------------
// GPT block for MI455X (gfx1250, wave32, WMMA + TDM). bf16 operands, fp32
// accum. Shapes fixed by the reference: B=8, S=1024, D=512, H=8, d=64.
// ---------------------------------------------------------------------------

typedef __attribute__((ext_vector_type(16))) __bf16 v16bf;
typedef __attribute__((ext_vector_type(8)))  __bf16 bf16x8;
typedef __attribute__((ext_vector_type(8)))  float  v8f;
typedef __attribute__((ext_vector_type(4)))  float  f32x4;
typedef __attribute__((ext_vector_type(4)))  unsigned int u32x4;
typedef __attribute__((ext_vector_type(8)))  unsigned int u32x8;

__device__ __forceinline__ v8f wmma_bf16(v16bf a, v16bf b, v8f c) {
  // D = A(16x32 bf16) * B(32x16 bf16) + C(16x16 f32)
  return __builtin_amdgcn_wmma_f32_16x16x32_bf16(false, a, false, b,
                                                 (short)0, c, false, false);
}

__device__ __forceinline__ v8f zero8() {
  v8f z;
#pragma unroll
  for (int i = 0; i < 8; ++i) z[i] = 0.0f;
  return z;
}

// ---------------------------------------------------------------------------
// TDM: DMA a 2-D tile (tile_w x tile_h elements, row stride = stride_elems)
// from global memory into LDS at byte offset lds_off. D# built per CDNA5 ISA
// CDNA5 ISA: group0 = {flags, lds_addr, global_addr, type=2},
// group1 = {data_size, tensor dims (huge, no OOB clip), tile dims, stride}.
// 2-D tensor -> groups 2/3 omitted (VADDR2/3 = NULL form).
// Tracked by TENSORcnt; caller waits with s_wait_tensorcnt.
// ---------------------------------------------------------------------------
template <int DSIZE_LOG2>  // 1: bf16, 2: fp32
__device__ __forceinline__ void tdm_load_2d(unsigned lds_off, const void* gaddr,
                                            unsigned tile_w, unsigned tile_h,
                                            unsigned stride_elems) {
  const unsigned long long ga = (unsigned long long)gaddr;
  u32x4 g0;
  g0[0] = 1u;                                    // count=1 (valid), load, user
  g0[1] = lds_off;                               // lds_addr (bytes)
  g0[2] = (unsigned)ga;                          // global_addr[31:0]
  g0[3] = (unsigned)((ga >> 32) & 0x01FFFFFFu)   // global_addr[56:32]
        | (2u << 30);                            // type = 2 ("image")
  u32x8 g1;
  g1[0] = (unsigned)DSIZE_LOG2 << 16;            // data_size; wg_mask=0
  g1[1] = 0u;                                    // tensor_dim0[15:0] (2^30)
  g1[2] = 0x4000u;                               // tensor_dim0 hi / dim1 lo
  g1[3] = 0x4000u | (tile_w << 16);              // tensor_dim1 hi, tile_dim0
  g1[4] = tile_h;                                // tile_dim1; tile_dim2 = 0
  g1[5] = stride_elems;                          // tensor_dim0_stride[31:0]
  g1[6] = 0u;                                    // stride hi, dim1_stride lo
  g1[7] = 0u;
  asm volatile("tensor_load_to_lds %0, %1" :: "s"(g0), "s"(g1) : "memory");
}

// A fragment (16x32, bf16) from a row-major bf16 matrix (global or LDS).
// Contiguous k order: lanes 0-15 -> M=lane, K = [k0..k0+16)
//             lanes 16-31 -> M=lane-16, K = [k0+16..k0+32)
__device__ __forceinline__ v16bf load_a_bf16(const __bf16* A, int lda,
                                             int row0, int k0, int lane) {
  const int r   = row0 + (lane & 15);
  const int off = (lane < 16) ? 0 : 16;
  const __bf16* p = A + (size_t)r * lda + k0 + off;
  bf16x8 lo = *(const bf16x8*)(p);
  bf16x8 hi = *(const bf16x8*)(p + 8);
  v16bf a;
#pragma unroll
  for (int j = 0; j < 8; ++j) { a[j] = lo[j]; a[j + 8] = hi[j]; }
  return a;
}

// A fragment from a row-major fp32 matrix, converted to bf16 in registers.
__device__ __forceinline__ v16bf load_a_f32(const float* A, int lda,
                                            int row0, int k0, int lane) {
  const int r   = row0 + (lane & 15);
  const int off = (lane < 16) ? 0 : 16;
  const float* p = A + (size_t)r * lda + k0 + off;
  f32x4 x0 = *(const f32x4*)(p);
  f32x4 x1 = *(const f32x4*)(p + 4);
  f32x4 x2 = *(const f32x4*)(p + 8);
  f32x4 x3 = *(const f32x4*)(p + 12);
  v16bf a;
#pragma unroll
  for (int j = 0; j < 4; ++j) {
    a[j]      = (__bf16)x0[j];
    a[j + 4]  = (__bf16)x1[j];
    a[j + 8]  = (__bf16)x2[j];
    a[j + 12] = (__bf16)x3[j];
  }
  return a;
}

// B fragment (32x16, bf16) from pre-packed tiles: lane's 16 values contiguous.
__device__ __forceinline__ v16bf load_b_pk(const __bf16* Bpk, size_t tile,
                                           int lane) {
  const __bf16* p = Bpk + tile * 512 + (size_t)lane * 16;
  bf16x8 lo = *(const bf16x8*)(p);
  bf16x8 hi = *(const bf16x8*)(p + 8);
  v16bf b;
#pragma unroll
  for (int j = 0; j < 8; ++j) { b[j] = lo[j]; b[j + 8] = hi[j]; }
  return b;
}

// B fragment for scores = Q * K^T: B[kd, kpos] = K[kpos, kd]. A lane's 16
// values are 16 *contiguous* d-values of one K row -> two b128 loads.
__device__ __forceinline__ v16bf load_b_kT(const __bf16* Km, int ldk,
                                           int n0, int k0, int lane) {
  const int r    = n0 + (lane & 15);
  const int koff = (lane < 16) ? 0 : 16;
  const __bf16* p = Km + (size_t)r * ldk + k0 + koff;
  bf16x8 lo = *(const bf16x8*)(p);
  bf16x8 hi = *(const bf16x8*)(p + 8);
  v16bf b;
#pragma unroll
  for (int j = 0; j < 8; ++j) { b[j] = lo[j]; b[j + 8] = hi[j]; }
  return b;
}

// ---------------------------------------------------------------------------
// Elementwise fp32 -> bf16
// ---------------------------------------------------------------------------
__global__ void f32_to_bf16_kernel(const float* __restrict__ x,
                                   __bf16* __restrict__ y, int n) {
  int i = blockIdx.x * blockDim.x + threadIdx.x;
  if (i < n) y[i] = (__bf16)x[i];
}

// ---------------------------------------------------------------------------
// Pack fp32 weight (K x N row-major) into WMMA B-fragment tile order.
// ---------------------------------------------------------------------------
__global__ void pack_w_f32_kernel(const float* __restrict__ W,
                                  __bf16* __restrict__ out, int K, int N) {
  const int gid  = blockIdx.x * blockDim.x + threadIdx.x;
  const int lane = gid & 31;
  const int tile = gid >> 5;
  const int Ktiles = K >> 5;
  const int total  = Ktiles * (N >> 4);
  if (tile >= total) return;
  const int nt = tile / Ktiles;
  const int kt = tile % Ktiles;
  const int col  = nt * 16 + (lane & 15);
  const int krow = kt * 32 + ((lane < 16) ? 0 : 16);
  __bf16* o = out + (size_t)tile * 512 + (size_t)lane * 16;
#pragma unroll
  for (int j = 0; j < 16; ++j)
    o[j] = (__bf16)W[(size_t)(krow + j) * N + col];
}

// Pack V (per (b,h): 1024x64 view, lda=512, bf16) into fragment tiles.
__global__ void pack_v_kernel(const __bf16* __restrict__ V,
                              __bf16* __restrict__ out) {
  const int bh = blockIdx.y;
  const __bf16* src = V + (size_t)(bh >> 3) * 1024 * 512 + (bh & 7) * 64;
  __bf16* dst = out + (size_t)bh * 128 * 512;  // 32 Ktiles * 4 Ntiles
  const int gid  = blockIdx.x * blockDim.x + threadIdx.x;
  const int lane = gid & 31;
  const int tile = gid >> 5;
  if (tile >= 128) return;
  const int nt = tile / 32;
  const int kt = tile % 32;
  const int col  = nt * 16 + (lane & 15);
  const int krow = kt * 32 + ((lane < 16) ? 0 : 16);
  __bf16* o = dst + (size_t)tile * 512 + (size_t)lane * 16;
#pragma unroll
  for (int j = 0; j < 16; ++j)
    o[j] = src[(size_t)(krow + j) * 512 + col];
}

// ---------------------------------------------------------------------------
// Generic GEMM: C[M,N] = A[M,K](bf16 row-major) * Bpk (packed bf16).
// A tiles (64 x 128, shared by all 4 waves of the block) are staged into LDS
// by the Tensor Data Mover, double-buffered (2 x 16 KB dynamic LDS), and
// consumed via ds loads. B fragments come straight from the packed weights
// (small, L2-resident). MODE 0: fp32 out; 1: relu->bf16; 2: bf16.
// ---------------------------------------------------------------------------
template <int MODE>
__global__ void gemm_bf16_pk_kernel(const __bf16* __restrict__ A, int lda,
                                    const __bf16* __restrict__ Bpk, int Ktiles,
                                    float* __restrict__ Cf,
                                    __bf16* __restrict__ Cb, int ldc) {
  extern __shared__ __bf16 smA[];  // 2 * 64*128 bf16 (offset 0: dynamic LDS)
  const int lane = threadIdx.x & 31;
  const int wave = threadIdx.x >> 5;
  const bool leader = (threadIdx.x < 32);
  const int m0 = blockIdx.x * 64;
  const int n0 = blockIdx.y * 128 + wave * 32;
  const int nt0 = n0 >> 4;
  const int CH = Ktiles >> 2;  // 128-column chunks

  v8f acc[4][2];
#pragma unroll
  for (int i = 0; i < 4; ++i) { acc[i][0] = zero8(); acc[i][1] = zero8(); }

  if (leader)
    tdm_load_2d<1>(0u, A + (size_t)m0 * lda, 128u, 64u, (unsigned)lda);

  for (int c = 0; c < CH; ++c) {
    if (leader) __builtin_amdgcn_s_wait_tensorcnt(0);
    __syncthreads();  // chunk c visible to every wave
    if (leader && c + 1 < CH)
      tdm_load_2d<1>((unsigned)(((c + 1) & 1) * 64 * 128 * 2),
                     A + (size_t)m0 * lda + (size_t)(c + 1) * 128,
                     128u, 64u, (unsigned)lda);
    if (c + 1 < CH) {  // warm L2/WGP$ for the next packed-B chunk
      __builtin_prefetch(Bpk + ((size_t)nt0 * Ktiles + (c + 1) * 4) * 512, 0, 1);
      __builtin_prefetch(Bpk + ((size_t)(nt0 + 1) * Ktiles + (c + 1) * 4) * 512, 0, 1);
    }
    const __bf16* Ab = smA + (c & 1) * (64 * 128);
#pragma unroll
    for (int ks = 0; ks < 4; ++ks) {
      const int kt = c * 4 + ks;
      const int k0 = ks << 5;
      v16bf a0 = load_a_bf16(Ab, 128,  0, k0, lane);
      v16bf a1 = load_a_bf16(Ab, 128, 16, k0, lane);
      v16bf a2 = load_a_bf16(Ab, 128, 32, k0, lane);
      v16bf a3 = load_a_bf16(Ab, 128, 48, k0, lane);
      v16bf b0 = load_b_pk(Bpk, (size_t)(nt0 + 0) * Ktiles + kt, lane);
      v16bf b1 = load_b_pk(Bpk, (size_t)(nt0 + 1) * Ktiles + kt, lane);
      acc[0][0] = wmma_bf16(a0, b0, acc[0][0]);
      acc[1][0] = wmma_bf16(a1, b0, acc[1][0]);
      acc[2][0] = wmma_bf16(a2, b0, acc[2][0]);
      acc[3][0] = wmma_bf16(a3, b0, acc[3][0]);
      acc[0][1] = wmma_bf16(a0, b1, acc[0][1]);
      acc[1][1] = wmma_bf16(a1, b1, acc[1][1]);
      acc[2][1] = wmma_bf16(a2, b1, acc[2][1]);
      acc[3][1] = wmma_bf16(a3, b1, acc[3][1]);
    }
    __syncthreads();  // all waves done with buffer (c&1) before reuse
  }

  const int rsub = (lane >> 4) << 3;  // lanes 16-31 hold rows M+8..M+15
  const int csub = lane & 15;
#pragma unroll
  for (int mi = 0; mi < 4; ++mi) {
#pragma unroll
    for (int ni = 0; ni < 2; ++ni) {
      const int col = n0 + ni * 16 + csub;
#pragma unroll
      for (int i = 0; i < 8; ++i) {
        const int row = m0 + mi * 16 + rsub + i;
        const float v = acc[mi][ni][i];
        if (MODE == 0)      Cf[(size_t)row * ldc + col] = v;
        else if (MODE == 1) Cb[(size_t)row * ldc + col] = (__bf16)fmaxf(v, 0.0f);
        else                Cb[(size_t)row * ldc + col] = (__bf16)v;
      }
    }
  }
}

// ---------------------------------------------------------------------------
// scores = (Q K^T)/8, causal mask, fp32 softmax, write probs to d_out.
// One 256-thread block per (b, h, 16 query rows). Row stride padded to 1036
// floats (1036 % 64 != 0 and 8*1036 % 64 != 0 -> conflict-free halves).
// ---------------------------------------------------------------------------
#define SM_STRIDE 1036
__global__ void attn_scores_softmax_kernel(const __bf16* __restrict__ Q,
                                           const __bf16* __restrict__ Km,
                                           float* __restrict__ Attn) {
  extern __shared__ float sm[];  // 16 * SM_STRIDE floats
  const int lane = threadIdx.x & 31;
  const int wave = threadIdx.x >> 5;
  const int qb = blockIdx.x;
  const int bh = blockIdx.y;
  const int b = bh >> 3, h = bh & 7;
  const __bf16* Qp = Q + (size_t)b * 1024 * 512 + h * 64;
  const __bf16* Kp = Km + (size_t)b * 1024 * 512 + h * 64;
  float* Ap = Attn + ((size_t)bh * 1024 + (size_t)qb * 16) * 1024;
  const int q0 = qb * 16;

  const v16bf aq0 = load_a_bf16(Qp, 512, q0, 0, lane);
  const v16bf aq1 = load_a_bf16(Qp, 512, q0, 32, lane);
  const int rsub = (lane >> 4) << 3;
  const int csub = lane & 15;

#pragma unroll
  for (int t = 0; t < 8; ++t) {
    const int n0 = (wave * 8 + t) * 16;
    v16bf bk0 = load_b_kT(Kp, 512, n0, 0, lane);
    v16bf bk1 = load_b_kT(Kp, 512, n0, 32, lane);
    v8f acc = zero8();
    acc = wmma_bf16(aq0, bk0, acc);
    acc = wmma_bf16(aq1, bk1, acc);
    const int col = n0 + csub;
#pragma unroll
    for (int i = 0; i < 8; ++i) {
      const int rloc = rsub + i;
      float s = acc[i] * 0.125f;                 // 1/sqrt(64)
      if (col > q0 + rloc) s = -1.0e9f;          // causal mask
      sm[rloc * SM_STRIDE + col] = s;
    }
  }
  __syncthreads();

#pragma unroll
  for (int rr = 0; rr < 2; ++rr) {
    const int r = wave * 2 + rr;
    float mx = -3.0e38f;
    for (int c = lane; c < 1024; c += 32) mx = fmaxf(mx, sm[r * SM_STRIDE + c]);
    for (int o = 16; o > 0; o >>= 1) mx = fmaxf(mx, __shfl_xor(mx, o, 32));
    float sum = 0.0f;
    for (int c = lane; c < 1024; c += 32) {
      const float e = __expf(sm[r * SM_STRIDE + c] - mx);
      sm[r * SM_STRIDE + c] = e;
      sum += e;
    }
    for (int o = 16; o > 0; o >>= 1) sum += __shfl_xor(sum, o, 32);
    const float inv = 1.0f / sum;
    for (int c = lane; c < 1024; c += 32)
      Ap[(size_t)r * 1024 + c] = sm[r * SM_STRIDE + c] * inv;
  }
}

// ---------------------------------------------------------------------------
// ctx = P * V per (b,h): M=1024, N=64, K=1024. The attention probabilities
// (268 MB, the dominant stream, bigger than L2) are staged 64x128-fp32 at a
// time into LDS by the TDM — one DMA per block instead of 4 redundant
// per-wave global reads — then converted to bf16 in registers. V comes from
// packed fragment tiles. 2 x 32 KB LDS double buffer.
// ---------------------------------------------------------------------------
__global__ void attn_ctx_kernel(const float* __restrict__ Attn,
                                const __bf16* __restrict__ Vpk,
                                __bf16* __restrict__ Ctx) {
  extern __shared__ float smP[];  // 2 * 64*128 fp32
  const int lane = threadIdx.x & 31;
  const int wave = threadIdx.x >> 5;  // n-tile 0..3
  const bool leader = (threadIdx.x < 32);
  const int bh = blockIdx.y;
  const int m0 = blockIdx.x * 64;
  const float* Ap = Attn + (size_t)bh * 1024 * 1024;
  const __bf16* Vp = Vpk + (size_t)bh * 128 * 512;

  v8f acc[4];
#pragma unroll
  for (int i = 0; i < 4; ++i) acc[i] = zero8();

  if (leader)
    tdm_load_2d<2>(0u, Ap + (size_t)m0 * 1024, 128u, 64u, 1024u);

  for (int c = 0; c < 8; ++c) {  // 8 chunks of 128 key positions
    if (leader) __builtin_amdgcn_s_wait_tensorcnt(0);
    __syncthreads();
    if (leader && c + 1 < 8)
      tdm_load_2d<2>((unsigned)(((c + 1) & 1) * 64 * 128 * 4),
                     Ap + (size_t)m0 * 1024 + (size_t)(c + 1) * 128,
                     128u, 64u, 1024u);
    const float* Pb = smP + (c & 1) * (64 * 128);
#pragma unroll
    for (int ks = 0; ks < 4; ++ks) {
      const int kt = c * 4 + ks;
      const int k0 = ks << 5;
      v16bf bv = load_b_pk(Vp, (size_t)wave * 32 + kt, lane);
      v16bf a0 = load_a_f32(Pb, 128,  0, k0, lane);
      v16bf a1 = load_a_f32(Pb, 128, 16, k0, lane);
      v16bf a2 = load_a_f32(Pb, 128, 32, k0, lane);
      v16bf a3 = load_a_f32(Pb, 128, 48, k0, lane);
      acc[0] = wmma_bf16(a0, bv, acc[0]);
      acc[1] = wmma_bf16(a1, bv, acc[1]);
      acc[2] = wmma_bf16(a2, bv, acc[2]);
      acc[3] = wmma_bf16(a3, bv, acc[3]);
    }
    __syncthreads();
  }

  const int b = bh >> 3, h = bh & 7;
  const int rsub = (lane >> 4) << 3;
  const int col = h * 64 + wave * 16 + (lane & 15);
#pragma unroll
  for (int mi = 0; mi < 4; ++mi) {
#pragma unroll
    for (int i = 0; i < 8; ++i) {
      const size_t row = (size_t)b * 1024 + m0 + mi * 16 + rsub + i;
      Ctx[row * 512 + col] = (__bf16)acc[mi][i];
    }
  }
}

// ---------------------------------------------------------------------------
// y = LayerNorm(x + res) * g + beta. One wave per 512-element row.
// ---------------------------------------------------------------------------
__global__ void ln_residual_kernel(const float* __restrict__ x,
                                   const float* __restrict__ res,
                                   const float* __restrict__ g,
                                   const float* __restrict__ beta,
                                   float* __restrict__ yf,
                                   __bf16* __restrict__ ybf) {
  const int lane = threadIdx.x & 31;
  const int wave = threadIdx.x >> 5;
  const size_t row = (size_t)blockIdx.x * 8 + wave;
  const float* xp = x + row * 512;
  const float* rp = res + row * 512;
  float vals[16];
  float s = 0.0f;
#pragma unroll
  for (int j = 0; j < 16; ++j) {
    const int c = lane + (j << 5);
    const float v = xp[c] + rp[c];
    vals[j] = v;
    s += v;
  }
  for (int o = 16; o > 0; o >>= 1) s += __shfl_xor(s, o, 32);
  const float mu = s * (1.0f / 512.0f);
  float vs = 0.0f;
#pragma unroll
  for (int j = 0; j < 16; ++j) { const float d = vals[j] - mu; vs += d * d; }
  for (int o = 16; o > 0; o >>= 1) vs += __shfl_xor(vs, o, 32);
  const float rstd = rsqrtf(vs * (1.0f / 512.0f) + 1e-5f);
#pragma unroll
  for (int j = 0; j < 16; ++j) {
    const int c = lane + (j << 5);
    const float y = (vals[j] - mu) * rstd * g[c] + beta[c];
    yf[row * 512 + c] = y;
    if (ybf) ybf[row * 512 + c] = (__bf16)y;
  }
}

// ---------------------------------------------------------------------------
// Launcher
// ---------------------------------------------------------------------------
extern "C" void kernel_launch(void* const* d_in, const int* in_sizes, int n_in,
                              void* d_out, int out_size, void* d_ws,
                              size_t ws_size, hipStream_t stream) {
  (void)in_sizes; (void)n_in; (void)out_size; (void)ws_size;
  const float* X  = (const float*)d_in[0];
  // d_in[1] = mask (bool) — causal mask is recomputed in-kernel.
  const float* Wq = (const float*)d_in[2];
  const float* Wk = (const float*)d_in[3];
  const float* Wv = (const float*)d_in[4];
  const float* Wo = (const float*)d_in[5];
  const float* g1 = (const float*)d_in[6];
  const float* b1 = (const float*)d_in[7];
  const float* W1 = (const float*)d_in[8];
  const float* W2 = (const float*)d_in[9];
  const float* g2 = (const float*)d_in[10];
  const float* b2 = (const float*)d_in[11];

  float* out_f = (float*)d_out;                            // [8,1024,512]
  float* attn  = out_f + (size_t)8 * 1024 * 512;           // [8,8,1024,1024]

  char* wp = (char*)d_ws;
  auto alloc = [&](size_t bytes) -> char* {
    char* p = wp;
    wp += (bytes + 255) & ~(size_t)255;
    return p;
  };
  const size_t BS = (size_t)8 * 1024;                      // rows
  __bf16* Xbf   = (__bf16*)alloc(BS * 512 * 2);
  __bf16* WqPk  = (__bf16*)alloc((size_t)512 * 512 * 2);
  __bf16* WkPk  = (__bf16*)alloc((size_t)512 * 512 * 2);
  __bf16* WvPk  = (__bf16*)alloc((size_t)512 * 512 * 2);
  __bf16* WoPk  = (__bf16*)alloc((size_t)512 * 512 * 2);
  __bf16* W1Pk  = (__bf16*)alloc((size_t)512 * 2048 * 2);
  __bf16* W2Pk  = (__bf16*)alloc((size_t)2048 * 512 * 2);
  __bf16* Qbf   = (__bf16*)alloc(BS * 512 * 2);
  __bf16* Kbf   = (__bf16*)alloc(BS * 512 * 2);
  __bf16* Vbf   = (__bf16*)alloc(BS * 512 * 2);
  __bf16* Vpk   = (__bf16*)alloc(BS * 512 * 2);
  __bf16* Ctxbf = (__bf16*)alloc(BS * 512 * 2);
  float*  TmpF  = (float*) alloc(BS * 512 * 4);            // O-proj, then FFN2
  float*  AOutF = (float*) alloc(BS * 512 * 4);            // attn_out fp32
  __bf16* AOutB = (__bf16*)alloc(BS * 512 * 2);            // attn_out bf16
  __bf16* H1bf  = (__bf16*)alloc(BS * 2048 * 2);           // relu(ffn hidden)

  const size_t gemm_lds = 2 * 64 * 128 * sizeof(__bf16);   // 32 KB
  const size_t ctx_lds  = 2 * 64 * 128 * sizeof(float);    // 64 KB
  const size_t sm_lds   = 16 * SM_STRIDE * sizeof(float);  // ~65 KB

  // 1) X -> bf16
  {
    const int n = (int)(BS * 512);
    f32_to_bf16_kernel<<<(n + 255) / 256, 256, 0, stream>>>(X, Xbf, n);
  }
  // 2) pack weights into fragment tile order
  auto packw = [&](const float* W, __bf16* O, int K, int N) {
    const int threads = (K >> 5) * (N >> 4) * 32;
    pack_w_f32_kernel<<<(threads + 255) / 256, 256, 0, stream>>>(W, O, K, N);
  };
  packw(Wq, WqPk, 512, 512);
  packw(Wk, WkPk, 512, 512);
  packw(Wv, WvPk, 512, 512);
  packw(Wo, WoPk, 512, 512);
  packw(W1, W1Pk, 512, 2048);
  packw(W2, W2Pk, 2048, 512);

  // 3) Q/K/V projections (M=8192, N=512, K=512), bf16 outputs
  {
    dim3 grid(8192 / 64, 512 / 128);
    gemm_bf16_pk_kernel<2><<<grid, 128, gemm_lds, stream>>>(Xbf, 512, WqPk, 16,
                                                            nullptr, Qbf, 512);
    gemm_bf16_pk_kernel<2><<<grid, 128, gemm_lds, stream>>>(Xbf, 512, WkPk, 16,
                                                            nullptr, Kbf, 512);
    gemm_bf16_pk_kernel<2><<<grid, 128, gemm_lds, stream>>>(Xbf, 512, WvPk, 16,
                                                            nullptr, Vbf, 512);
  }
  // 4) pack V into per-(b,h) fragment tiles
  {
    dim3 grid((128 * 32 + 255) / 256, 64);
    pack_v_kernel<<<grid, 256, 0, stream>>>(Vbf, Vpk);
  }
  // 5) scores + causal mask + softmax -> attention probs (fp32, into d_out)
  {
    dim3 grid(1024 / 16, 64);
    attn_scores_softmax_kernel<<<grid, 256, sm_lds, stream>>>(Qbf, Kbf, attn);
  }
  // 6) ctx = P * V (bf16 output, [B,S,512]); TDM-staged probabilities
  {
    dim3 grid(1024 / 64, 64);
    attn_ctx_kernel<<<grid, 128, ctx_lds, stream>>>(attn, Vpk, Ctxbf);
  }
  // 7) O-projection (fp32 out) then LN1 with residual = X
  {
    dim3 grid(8192 / 64, 512 / 128);
    gemm_bf16_pk_kernel<0><<<grid, 128, gemm_lds, stream>>>(Ctxbf, 512, WoPk, 16,
                                                            TmpF, nullptr, 512);
    ln_residual_kernel<<<8192 / 8, 256, 0, stream>>>(TmpF, X, g1, b1,
                                                     AOutF, AOutB);
  }
  // 8) FFN: H1 = relu(attn_out @ W1) -> bf16 (N=2048)
  {
    dim3 grid(8192 / 64, 2048 / 128);
    gemm_bf16_pk_kernel<1><<<grid, 128, gemm_lds, stream>>>(AOutB, 512, W1Pk, 16,
                                                            nullptr, H1bf, 2048);
  }
  // 9) F = H1 @ W2 (K=2048, N=512, fp32 out) then LN2 -> d_out
  {
    dim3 grid(8192 / 64, 512 / 128);
    gemm_bf16_pk_kernel<0><<<grid, 128, gemm_lds, stream>>>(H1bf, 2048, W2Pk, 64,
                                                            TmpF, nullptr, 512);
    ln_residual_kernel<<<8192 / 8, 256, 0, stream>>>(TmpF, AOutF, g2, b2,
                                                     out_f, nullptr);
  }
}
